// GlGraphConv_32220844655420
// MI455X (gfx1250) — matrix operand order repaired and checked by reading, not compiled, not run on hardware
//
#include <hip/hip_runtime.h>
#include <math.h>

typedef __attribute__((ext_vector_type(16))) __bf16 v16bf;
typedef __attribute__((ext_vector_type(8)))  float  v8f;

#define NEGV (-9.0e15f)

// ---- bf16 helpers (round-to-nearest-even) ------------------------------
__device__ __forceinline__ unsigned bf16pack(float a, float b) {
    unsigned ua = __builtin_bit_cast(unsigned, a);
    unsigned ub = __builtin_bit_cast(unsigned, b);
    unsigned ra = (ua + (0x7FFFu + ((ua >> 16) & 1u))) >> 16;
    unsigned rb = (ub + (0x7FFFu + ((ub >> 16) & 1u))) >> 16;
    return (ra & 0xFFFFu) | (rb << 16);
}
__device__ __forceinline__ float bf16hi_val(float a) {
    unsigned ua = __builtin_bit_cast(unsigned, a);
    unsigned ra = (ua + (0x7FFFu + ((ua >> 16) & 1u))) & 0xFFFF0000u;
    return __builtin_bit_cast(float, ra);
}

// ---- prep: softmaxed adjacency (sparse chain) + softmax(MM) ------------
// wsmall layout (floats): [0..15] selfScale, [16..31] aL, [32..47] aR,
//                         [48..175] fdc[16][8]
__global__ void glgc_prep(const float* __restrict__ e,
                          const float* __restrict__ MM,
                          const int* __restrict__ mr,
                          const int* __restrict__ mc,
                          int nnz, float* __restrict__ wsmall) {
    __shared__ float full[16][16];
    const int tid = threadIdx.x;
    {
        int r = tid >> 4, c = tid & 15;
        float v = NEGV;
        for (int j = 0; j < nnz; ++j)
            if (mr[j] == r && mc[j] == c) v = e[j];
        full[r][c] = v;
    }
    __syncthreads();
    if (tid < 16) {
        const int n = tid;
        float m = full[n][0];
        for (int j = 1; j < 16; ++j) m = fmaxf(m, full[n][j]);
        float s = 0.f;
        for (int j = 0; j < 16; ++j) s += expf(full[n][j] - m);
        float inv = 1.0f / s;
        wsmall[n]      = expf(full[n][n] - m) * inv;                      // diag
        wsmall[16 + n] = (n > 0)  ? expf(full[n][n - 1] - m) * inv : 0.f; // left
        wsmall[32 + n] = (n < 15) ? expf(full[n][n + 1] - m) * inv : 0.f; // right
        const float* mmr = MM + n * 8;
        float mx = mmr[0];
        for (int j = 1; j < 8; ++j) mx = fmaxf(mx, mmr[j]);
        float ss = 0.f;
        for (int j = 0; j < 8; ++j) ss += expf(mmr[j] - mx);
        float iv = 1.0f / ss;
        for (int j = 0; j < 8; ++j) wsmall[48 + n * 8 + j] = expf(mmr[j] - mx) * iv;
    }
}

// ---- pack W into bf16 hi/lo B-operand tiles ----------------------------
// dword gid = ((((k*8+t)*4+q)*2+part)*32 + lane)*8 + v
// lane<16: col N=16t+lane, K=32q+2v..; lane>=16: K += 16
__global__ void glgc_packw(const float* __restrict__ W, unsigned* __restrict__ wp) {
    int gid = blockIdx.x * 256 + threadIdx.x;
    if (gid >= 3 * 8 * 4 * 2 * 32 * 8) return;
    int v    =  gid        & 7;
    int lane = (gid >> 3)  & 31;
    int part = (gid >> 8)  & 1;
    int q    = (gid >> 9)  & 3;
    int t    = (gid >> 11) & 7;
    int k    =  gid >> 14;
    int col = lane & 15, half = lane >> 4;
    int N = t * 16 + col;
    int K = q * 32 + half * 16 + 2 * v;
    float f0 = W[(k * 128 + K) * 128 + N];
    float f1 = W[(k * 128 + K + 1) * 128 + N];
    if (part) { f0 -= bf16hi_val(f0); f1 -= bf16hi_val(f1); }
    wp[gid] = bf16pack(f0, f1);
}

// ---- main fused kernel: one sample per block, one 16-col tile per wave --
__global__ __launch_bounds__(256, 2)
void glgc_main(const float* __restrict__ x, const float* __restrict__ bias,
               const float* __restrict__ wsmall, const unsigned* __restrict__ wpak,
               float* __restrict__ out) {
    __shared__ __align__(16) unsigned sxh[1024];   // bf16-hi pairs, [row][64]
    __shared__ __align__(16) unsigned sxl[1024];   // bf16-lo pairs
    __shared__ float sp[176];

    const int tid = threadIdx.x;
    const int sample = blockIdx.x;

    // stage x: 2048 floats -> packed bf16 hi/lo pairs (A-operand layout)
    const float4* xs = (const float4*)(x + (size_t)sample * 2048);
    float4 f0 = xs[tid * 2 + 0];
    float4 f1 = xs[tid * 2 + 1];
    uint4 hi4, lo4;
    hi4.x = bf16pack(f0.x, f0.y); hi4.y = bf16pack(f0.z, f0.w);
    hi4.z = bf16pack(f1.x, f1.y); hi4.w = bf16pack(f1.z, f1.w);
    lo4.x = bf16pack(f0.x - bf16hi_val(f0.x), f0.y - bf16hi_val(f0.y));
    lo4.y = bf16pack(f0.z - bf16hi_val(f0.z), f0.w - bf16hi_val(f0.w));
    lo4.z = bf16pack(f1.x - bf16hi_val(f1.x), f1.y - bf16hi_val(f1.y));
    lo4.w = bf16pack(f1.z - bf16hi_val(f1.z), f1.w - bf16hi_val(f1.w));
    ((uint4*)sxh)[tid] = hi4;
    ((uint4*)sxl)[tid] = lo4;
    if (tid < 176) sp[tid] = wsmall[tid];
    __syncthreads();

    const int lane = tid & 31;
    const int wave = tid >> 5;        // column tile t
    const int half = lane >> 4;
    const int row  = lane & 15;       // A row / D column
    const int col  = lane & 15;

    v8f c[3] = {};

    #pragma unroll
    for (int q = 0; q < 4; ++q) {
        const int b4 = (row * 64 + q * 16 + half * 8) >> 2;  // uint4 index
        union { uint4 u[2]; v16bf v; } ah, al;
        ah.u[0] = ((const uint4*)sxh)[b4];
        ah.u[1] = ((const uint4*)sxh)[b4 + 1];
        al.u[0] = ((const uint4*)sxl)[b4];
        al.u[1] = ((const uint4*)sxl)[b4 + 1];
        #pragma unroll
        for (int k = 0; k < 3; ++k) {
            const int bi = ((k * 8 + wave) * 4 + q) * 2;     // hi block
            const uint4* bph = (const uint4*)wpak + ((size_t)bi * 64 + lane * 2);
            const uint4* bpl = (const uint4*)wpak + ((size_t)(bi + 1) * 64 + lane * 2);
            union { uint4 u[2]; v16bf v; } bh, bl;
            bh.u[0] = bph[0]; bh.u[1] = bph[1];
            bl.u[0] = bpl[0]; bl.u[1] = bpl[1];
            c[k] = __builtin_amdgcn_wmma_f32_16x16x32_bf16(
                       false, ah.v, false, bh.v, (short)0, c[k], false, false);
            c[k] = __builtin_amdgcn_wmma_f32_16x16x32_bf16(
                       false, ah.v, false, bl.v, (short)0, c[k], false, false);
            c[k] = __builtin_amdgcn_wmma_f32_16x16x32_bf16(
                       false, al.v, false, bh.v, (short)0, c[k], false, false);
        }
    }

    // ---- fused epilogue in D layout (lane: col, rows r+8*half) ----------
    float h1r[16], h2r[16];
    #pragma unroll
    for (int r = 0; r < 8; ++r) {
        float o1 = c[1][r], s1 = __shfl_xor(o1, 16, 32);
        float o2 = c[2][r], s2 = __shfl_xor(o2, 16, 32);
        h1r[r]     = half ? s1 : o1;
        h1r[r + 8] = half ? o1 : s1;
        h2r[r]     = half ? s2 : o2;
        h2r[r + 8] = half ? o2 : s2;
    }
    float p[8];
    #pragma unroll
    for (int j = 0; j < 8; ++j) p[j] = fmaxf(h2r[2 * j], h2r[2 * j + 1]);

    const float bcol = bias[wave * 16 + col];
    float* ob = out + (size_t)sample * 2048 + wave * 16 + col;

#define GLGC_EMIT(N, R)                                                         \
    {                                                                           \
        float b0v = sp[N] * c[0][R];                                            \
        float b1v = sp[16 + N] * h1r[(N + 15) & 15] +                           \
                    sp[32 + N] * h1r[(N + 1) & 15];                             \
        float b3v = sp[48 + N * 8 + 0] * p[0] + sp[48 + N * 8 + 1] * p[1] +     \
                    sp[48 + N * 8 + 2] * p[2] + sp[48 + N * 8 + 3] * p[3] +     \
                    sp[48 + N * 8 + 4] * p[4] + sp[48 + N * 8 + 5] * p[5] +     \
                    sp[48 + N * 8 + 6] * p[6] + sp[48 + N * 8 + 7] * p[7];      \
        ob[(N) * 128] = fmaxf(fmaxf(b0v, b1v), b3v) + bcol;                     \
    }

    if (half == 0) {
        GLGC_EMIT(0, 0) GLGC_EMIT(1, 1) GLGC_EMIT(2, 2) GLGC_EMIT(3, 3)
        GLGC_EMIT(4, 4) GLGC_EMIT(5, 5) GLGC_EMIT(6, 6) GLGC_EMIT(7, 7)
    } else {
        GLGC_EMIT(8, 0)  GLGC_EMIT(9, 1)  GLGC_EMIT(10, 2) GLGC_EMIT(11, 3)
        GLGC_EMIT(12, 4) GLGC_EMIT(13, 5) GLGC_EMIT(14, 6) GLGC_EMIT(15, 7)
    }
#undef GLGC_EMIT
}

extern "C" void kernel_launch(void* const* d_in, const int* in_sizes, int n_in,
                              void* d_out, int out_size, void* d_ws, size_t ws_size,
                              hipStream_t stream) {
    const float* x    = (const float*)d_in[0];
    const float* W    = (const float*)d_in[1];
    const float* MM   = (const float*)d_in[2];
    const float* e    = (const float*)d_in[3];
    const float* bias = (const float*)d_in[4];
    const int*   mr   = (const int*)d_in[5];
    const int*   mc   = (const int*)d_in[6];

    float*    wsmall = (float*)d_ws;
    unsigned* wpak   = (unsigned*)((char*)d_ws + 1024);
    const int nnz = in_sizes[3];
    const int batch = in_sizes[0] / (16 * 128);

    glgc_prep<<<1, 256, 0, stream>>>(e, MM, mr, mc, nnz, wsmall);
    glgc_packw<<<192, 256, 0, stream>>>(W, wpak);
    glgc_main<<<batch, 256, 0, stream>>>(x, bias, wsmall, wpak, (float*)d_out);
}
